// Layer0AttentionWrapper_49478023250129
// MI455X (gfx1250) — matrix operand order repaired and checked by reading, not compiled, not run on hardware
//
#include <hip/hip_runtime.h>
#include <hip/hip_bf16.h>
#include <cstdint>

#define HIDDEN 2048
#define NH 16
#define NKV 2
#define HD 128
#define CTX 4096
#define NL 36
#define GQA 8          // NH / NKV
#define NSPLIT 32      // flash-decode splits per kv group
#define CHUNK 32       // positions per inner chunk (two 16-wide WMMA tiles)
#define AW 4           // waves per attention block

typedef __attribute__((ext_vector_type(16))) _Float16 v16h;
typedef __attribute__((ext_vector_type(8)))  float    v8f;
typedef __attribute__((ext_vector_type(4)))  float    vf4;

// gfx1250 async global->LDS copy (16B per lane), tracked by ASYNCcnt.
// vdst = 32-bit LDS byte address, vaddr = 64-bit global address.
#define ASYNC_CP16(gsrc, ldst)                                              \
    asm volatile("global_load_async_to_lds_b128 %0, %1, off"                \
                 :: "v"((unsigned)(uintptr_t)(ldst)),                       \
                    "v"((const float*)(gsrc)) : "memory")
#define WAIT_ASYNCCNT(N)                                                    \
    asm volatile("s_wait_asynccnt %0" :: "i"(N) : "memory")

// ---------------------------------------------------------------------------
// Kernel 1: RMSNorm  t[i] = x[i] * rsqrt(mean(x^2)+eps) * ln_w[i]
// ---------------------------------------------------------------------------
__global__ void k_rmsnorm(const float* __restrict__ x,
                          const float* __restrict__ w,
                          float* __restrict__ t) {
    __shared__ float red[8];
    const int tid = threadIdx.x;      // 256 threads
    float s = 0.f;
    for (int i = tid; i < HIDDEN; i += 256) { float v = x[i]; s += v * v; }
    #pragma unroll
    for (int off = 16; off; off >>= 1) s += __shfl_xor(s, off, 32);
    if ((tid & 31) == 0) red[tid >> 5] = s;
    __syncthreads();
    if (tid == 0) {
        float a = 0.f;
        #pragma unroll
        for (int i = 0; i < 8; ++i) a += red[i];
        red[0] = a;
    }
    __syncthreads();
    const float rs = rsqrtf(red[0] / (float)HIDDEN + 1e-6f);
    for (int i = tid; i < HIDDEN; i += 256) t[i] = x[i] * rs * w[i];
}

// ---------------------------------------------------------------------------
// Kernel 2: fused QKV GEMV (wave per row, float4 coalesced) -> qkv[2560]
// ---------------------------------------------------------------------------
__global__ void k_qkv_gemv(const float* __restrict__ wq, const float* __restrict__ bq,
                           const float* __restrict__ wk, const float* __restrict__ bk,
                           const float* __restrict__ wv, const float* __restrict__ bv,
                           const float* __restrict__ t,  float* __restrict__ out) {
    const int wid  = threadIdx.x >> 5;
    const int lane = threadIdx.x & 31;
    const int row  = blockIdx.x * 8 + wid;              // 0..2559
    const float* W; const float* B; int r;
    if (row < NH * HD)            { W = wq; B = bq; r = row; }
    else if (row < NH*HD + NKV*HD){ W = wk; B = bk; r = row - NH*HD; }
    else                          { W = wv; B = bv; r = row - NH*HD - NKV*HD; }
    const float* wr = W + (size_t)r * HIDDEN;
    float acc = 0.f;
    for (int i = lane * 4; i < HIDDEN; i += 128) {
        vf4 a = *(const vf4*)(wr + i);
        vf4 b = *(const vf4*)(t  + i);
        acc += a[0]*b[0] + a[1]*b[1] + a[2]*b[2] + a[3]*b[3];
    }
    #pragma unroll
    for (int off = 16; off; off >>= 1) acc += __shfl_xor(acc, off, 32);
    if (lane == 0) out[row] = acc + B[r];
}

// ---------------------------------------------------------------------------
// Kernel 3: RoPE on q (16 heads) and k (2 heads); v copied through.
// ---------------------------------------------------------------------------
__global__ void k_rope(const float* __restrict__ qkv,
                       const int*   __restrict__ curp,
                       float* __restrict__ qkvr) {
    const int tid = threadIdx.x + blockIdx.x * blockDim.x;
    const float pos = (float)(*curp);
    if (tid < (NH + NKV) * (HD / 2)) {          // 1152 rotation pairs
        const int head = tid >> 6;
        const int j    = tid & 63;
        const int off  = head * HD;
        const float inv = __expf(-((float)(2 * j) / (float)HD) * 13.815510557964274f);
        const float ang = pos * inv;
        float s, c;
        __sincosf(ang, &s, &c);
        const float x1 = qkv[off + j];
        const float x2 = qkv[off + j + 64];
        qkvr[off + j]      = x1 * c - x2 * s;
        qkvr[off + j + 64] = x2 * c + x1 * s;
    } else if (tid < (NH + NKV) * (HD / 2) + NKV * HD) {
        const int i = tid - (NH + NKV) * (HD / 2);
        qkvr[(NH + NKV) * HD + i] = qkv[(NH + NKV) * HD + i];   // v passthrough
    }
}

// ---------------------------------------------------------------------------
// Kernel 4: split-K flash-decode attention with WMMA f16 and async-to-LDS
// double-buffered K/V tiles. Per 32-position chunk:
//   scores: D[pos][head] = Ktile(16x32 f16) x Q^T(32x16 f16), f32 acc
//   ctx:    O[head][dim] += P(16x32 f16) x Vtile(32x16 f16), f32 acc
// ---------------------------------------------------------------------------
__global__ void __launch_bounds__(AW * 32)
k_attn_partial(const float* __restrict__ kvc,
               const float* __restrict__ qkvr,
               const int*   __restrict__ curp,
               float* __restrict__ pm,
               float* __restrict__ ps,
               float* __restrict__ po) {
    __shared__ float ldsK[AW][2][CHUNK * HD];   // 128 KB
    __shared__ float ldsV[AW][2][CHUNK * HD];   // 128 KB
    __shared__ v16h  ldsQ[4][32];               // 4 KB formatted Q B-matrices

    const int cur   = *curp;
    const int lane  = threadIdx.x & 31;
    const int wid   = threadIdx.x >> 5;         // 0..3
    const int wsid  = blockIdx.x * AW + wid;    // 0..63
    const int g     = wsid >> 5;                // kv group (uniform per block)
    const int split = wsid & (NSPLIT - 1);
    const bool hi   = lane >= 16;
    const int  n    = lane & 15;
    const float scale = 0.08838834764831845f;   // 1/sqrt(128)

    const float* keyc = kvc + (size_t)g * CTX * HD;                 // kv[0][g]
    const float* valc = kvc + ((size_t)NL * NKV + g) * CTX * HD;    // kv[NL][g]
    const float* knew = qkvr + NH * HD + g * HD;
    const float* vnew = qkvr + NH * HD + NKV * HD + g * HD;
    const float* qg   = qkvr + g * GQA * HD;

    // ---- cooperative build of formatted Q (B-matrix layout), shared per block
    {
        const int l  = threadIdx.x & 31;
        const int kb = threadIdx.x >> 5;        // 0..3
        const bool hl = (l & 16) != 0;
        const int  nl = l & 15;
        v16h q;
        #pragma unroll
        for (int i = 0; i < 8; ++i) {
            const int k0 = kb * 32 + (hl ? 8 : 0) + (i < 4 ? 2 * i : 16 + 2 * (i - 4));
            float lo = 0.f, hu = 0.f;
            if (nl < GQA) { lo = qg[nl * HD + k0]; hu = qg[nl * HD + k0 + 1]; }
            q[2 * i]     = (_Float16)lo;
            q[2 * i + 1] = (_Float16)hu;
        }
        ldsQ[kb][l] = q;
    }
    __syncthreads();

    const v8f zero = {0.f, 0.f, 0.f, 0.f, 0.f, 0.f, 0.f, 0.f};
    v8f o[8];
    #pragma unroll
    for (int i = 0; i < 8; ++i) o[i] = zero;
    float mx = -1e30f, sm = 0.f;

    const int cmax = cur / CHUNK;               // last chunk containing valid pos
    int nb = 0;

    // preload K tile for first chunk (contiguous 16KB, 32 async b128)
    if (split <= cmax) {
        const float* gk = keyc + (size_t)split * CHUNK * HD + lane * 4;
        #pragma unroll
        for (int j = 0; j < CHUNK; ++j)
            ASYNC_CP16(gk + j * HD, &ldsK[wid][0][j * HD + lane * 4]);
    }

    for (int c = split; c <= cmax; c += NSPLIT, nb ^= 1) {
        const int base = c * CHUNK;

        // issue V tile for this chunk (after K_c -> in-order completion)
        {
            const float* gv = valc + (size_t)base * HD + lane * 4;
            #pragma unroll
            for (int j = 0; j < CHUNK; ++j)
                ASYNC_CP16(gv + j * HD, &ldsV[wid][nb][j * HD + lane * 4]);
        }

        WAIT_ASYNCCNT(32);                      // K_c landed (V_c may still fly)

        // patch the current position's row with the fresh RoPE'd k
        if (cur >= base && cur < base + CHUNK) {
            const int r = cur - base;
            *(vf4*)&ldsK[wid][nb][r * HD + lane * 4] = *(const vf4*)(knew + lane * 4);
        }

        // ---- scores over two 16-position tiles ----
        v8f sT[2]; sT[0] = zero; sT[1] = zero;
        #pragma unroll
        for (int t = 0; t < 2; ++t) {
            const float* src = &ldsK[wid][nb][(t * 16 + n) * HD];
            #pragma unroll
            for (int kb = 0; kb < 4; ++kb) {
                const int k0 = kb * 32 + (hi ? 8 : 0);
                vf4 q0 = *(const vf4*)(src + k0);
                vf4 q1 = *(const vf4*)(src + k0 + 4);
                vf4 q2 = *(const vf4*)(src + k0 + 16);
                vf4 q3 = *(const vf4*)(src + k0 + 20);
                v16h a;
                a[0]=(_Float16)q0[0]; a[1]=(_Float16)q0[1]; a[2]=(_Float16)q0[2]; a[3]=(_Float16)q0[3];
                a[4]=(_Float16)q1[0]; a[5]=(_Float16)q1[1]; a[6]=(_Float16)q1[2]; a[7]=(_Float16)q1[3];
                a[8]=(_Float16)q2[0]; a[9]=(_Float16)q2[1]; a[10]=(_Float16)q2[2]; a[11]=(_Float16)q2[3];
                a[12]=(_Float16)q3[0]; a[13]=(_Float16)q3[1]; a[14]=(_Float16)q3[2]; a[15]=(_Float16)q3[3];
                v16h bqk = ldsQ[kb][lane];
                sT[t] = __builtin_amdgcn_wmma_f32_16x16x32_f16(
                            false, a, false, bqk, (short)0, sT[t], false, false);
            }
        }

        // ---- scale + causal mask + online softmax (per head = lane pair) ----
        float cm = -1e30f;
        #pragma unroll
        for (int t = 0; t < 2; ++t)
            #pragma unroll
            for (int r = 0; r < 8; ++r) {
                const int p = base + t * 16 + r + (hi ? 8 : 0);
                float v = sT[t][r] * scale;
                v = (p <= cur) ? v : -1e30f;
                sT[t][r] = v;
                cm = fmaxf(cm, v);
            }
        cm = fmaxf(cm, __shfl_xor(cm, 16, 32));
        const float nm   = fmaxf(mx, cm);
        const float fold = __expf(mx - nm);
        mx = nm;

        float psl = 0.f;
        #pragma unroll
        for (int t = 0; t < 2; ++t)
            #pragma unroll
            for (int r = 0; r < 8; ++r) {
                const float e = __expf(sT[t][r] - nm);
                sT[t][r] = e;
                psl += e;
            }
        psl += __shfl_xor(psl, 16, 32);
        sm = sm * fold + psl;

        #pragma unroll
        for (int r = 0; r < 8; ++r) {
            const float f = __shfl(fold, r + (hi ? 8 : 0), 32);
            #pragma unroll
            for (int dc = 0; dc < 8; ++dc) o[dc][r] *= f;
        }

        WAIT_ASYNCCNT(0);                       // V_c landed

        // patch the current position's row with the fresh v
        if (cur >= base && cur < base + CHUNK) {
            const int r = cur - base;
            *(vf4*)&ldsV[wid][nb][r * HD + lane * 4] = *(const vf4*)(vnew + lane * 4);
        }

        // prefetch next chunk's K tile; overlaps the ctx WMMAs below
        {
            const int cn = c + NSPLIT;
            if (cn <= cmax) {
                const float* gk = keyc + (size_t)cn * CHUNK * HD + lane * 4;
                #pragma unroll
                for (int j = 0; j < CHUNK; ++j)
                    ASYNC_CP16(gk + j * HD, &ldsK[wid][nb ^ 1][j * HD + lane * 4]);
            }
        }

        // ---- pack probabilities into f16 A-matrix (exact layout match) ----
        v16h ap;
        #pragma unroll
        for (int i = 0; i < 4; ++i) {
            ap[2*i]       = (_Float16)sT[0][2*i];
            ap[2*i + 1]   = (_Float16)sT[0][2*i + 1];
            ap[8 + 2*i]   = (_Float16)sT[1][2*i];
            ap[8 + 2*i+1] = (_Float16)sT[1][2*i + 1];
        }

        // ---- ctx accumulate: O[head][dim] += P x Vtile ----
        #pragma unroll
        for (int dc = 0; dc < 8; ++dc) {
            const int d = dc * 16 + n;
            v16h bv;
            #pragma unroll
            for (int i = 0; i < 8; ++i) {
                const int r0 = (hi ? 8 : 0) + (i < 4 ? 2 * i : 16 + 2 * (i - 4));
                bv[2*i]     = (_Float16)ldsV[wid][nb][r0 * HD + d];
                bv[2*i + 1] = (_Float16)ldsV[wid][nb][(r0 + 1) * HD + d];
            }
            o[dc] = __builtin_amdgcn_wmma_f32_16x16x32_f16(
                        false, ap, false, bv, (short)0, o[dc], false, false);
        }
    }

    // ---- write partials (valid heads only: rows 0..7, lanes 0..15) ----
    if (!hi) {
        if (n < GQA) {
            pm[(g * NSPLIT + split) * GQA + n] = mx;
            ps[(g * NSPLIT + split) * GQA + n] = sm;
        }
        #pragma unroll
        for (int dc = 0; dc < 8; ++dc)
            #pragma unroll
            for (int r = 0; r < 8; ++r)
                po[((size_t)(g * NSPLIT + split) * GQA + r) * HD + dc * 16 + n] = o[dc][r];
    }
}

// ---------------------------------------------------------------------------
// Kernel 5: combine split partials (log-sum-exp merge) -> ctx[16][128]
// ---------------------------------------------------------------------------
__global__ void k_attn_reduce(const float* __restrict__ pm,
                              const float* __restrict__ ps,
                              const float* __restrict__ po,
                              float* __restrict__ ctx) {
    const int h = blockIdx.x;        // 0..15
    const int d = threadIdx.x;       // 0..127
    const int g = h / GQA, nn = h % GQA;
    float M = -1e30f;
    for (int s = 0; s < NSPLIT; ++s)
        M = fmaxf(M, pm[(g * NSPLIT + s) * GQA + nn]);
    float tot = 0.f, acc = 0.f;
    for (int s = 0; s < NSPLIT; ++s) {
        const float w = __expf(pm[(g * NSPLIT + s) * GQA + nn] - M);
        tot += ps[(g * NSPLIT + s) * GQA + nn] * w;
        acc += po[((size_t)(g * NSPLIT + s) * GQA + nn) * HD + d] * w;
    }
    ctx[h * HD + d] = acc / tot;
}

// ---------------------------------------------------------------------------
// Kernel 6: output GEMV + residual  out[i] = hidden[i] + wo[i,:]·ctx
// ---------------------------------------------------------------------------
__global__ void k_out_gemv(const float* __restrict__ wo,
                           const float* __restrict__ ctx,
                           const float* __restrict__ hidden,
                           float* __restrict__ out) {
    const int wid  = threadIdx.x >> 5;
    const int lane = threadIdx.x & 31;
    const int row  = blockIdx.x * 8 + wid;     // 0..2047
    const float* wr = wo + (size_t)row * (NH * HD);
    float acc = 0.f;
    for (int i = lane * 4; i < NH * HD; i += 128) {
        vf4 a = *(const vf4*)(wr + i);
        vf4 b = *(const vf4*)(ctx + i);
        acc += a[0]*b[0] + a[1]*b[1] + a[2]*b[2] + a[3]*b[3];
    }
    #pragma unroll
    for (int off = 16; off; off >>= 1) acc += __shfl_xor(acc, off, 32);
    if (lane == 0) out[row] = hidden[row] + acc;
}

// ---------------------------------------------------------------------------
extern "C" void kernel_launch(void* const* d_in, const int* in_sizes, int n_in,
                              void* d_out, int out_size, void* d_ws, size_t ws_size,
                              hipStream_t stream) {
    const float* hidden = (const float*)d_in[0];
    const int*   curp   = (const int*)  d_in[3];
    const float* ln_w   = (const float*)d_in[4];
    const float* wq     = (const float*)d_in[5];
    const float* bq     = (const float*)d_in[6];
    const float* wk     = (const float*)d_in[7];
    const float* bk     = (const float*)d_in[8];
    const float* wv     = (const float*)d_in[9];
    const float* bv     = (const float*)d_in[10];
    const float* wo     = (const float*)d_in[11];
    const float* kvc    = (const float*)d_in[12];

    float* ws   = (float*)d_ws;
    float* t    = ws;            // 2048
    float* qkv  = ws + 2048;     // 2560 raw q|k|v
    float* qkvr = ws + 4608;     // 2560 rope'd q|k|v
    float* ctx  = ws + 7168;     // 2048
    float* pm   = ws + 9216;     // 512
    float* ps   = ws + 9728;     // 512
    float* po   = ws + 10240;    // 65536

    k_rmsnorm     <<<1,   256, 0, stream>>>(hidden, ln_w, t);
    k_qkv_gemv    <<<320, 256, 0, stream>>>(wq, bq, wk, bk, wv, bv, t, qkv);
    k_rope        <<<3,   512, 0, stream>>>(qkv, curp, qkvr);
    k_attn_partial<<<16,  AW * 32, 0, stream>>>(kvc, qkvr, curp, pm, ps, po);
    k_attn_reduce <<<16,  128, 0, stream>>>(pm, ps, po, ctx);
    k_out_gemv    <<<256, 256, 0, stream>>>(wo, ctx, hidden, (float*)d_out);
}
